// FlatGNN_52879637348374
// MI455X (gfx1250) — compile-verified, matrix operand order checked by reading, not run on hardware
//
#include <hip/hip_runtime.h>
#include <hip/hip_bf16.h>

#define N_NODES 50000
#define N_GRAPHS 256
#define HEADS 4
#define HID 64
#define CH 256            // HEADS*HID
#define OUT_DIM 128
#define NEG_SLOPE 0.2f

typedef __attribute__((ext_vector_type(16))) __bf16 v16bf;
typedef __attribute__((ext_vector_type(8)))  float  v8f;

__device__ __forceinline__ unsigned short f2bf(float f) {
    union { float f; unsigned int u; } c; c.f = f;
    unsigned int u = c.u;
    return (unsigned short)((u + 0x7FFFu + ((u >> 16) & 1u)) >> 16);
}

// ---------------- utility fills / converts ----------------
__global__ void fill_f32_kernel(float* p, float v, int n) {
    int i = blockIdx.x * blockDim.x + threadIdx.x;
    if (i < n) p[i] = v;
}

__global__ void cvt_bf16_kernel(const float* __restrict__ src,
                                unsigned short* __restrict__ dst, int n) {
    int i = blockIdx.x * blockDim.x + threadIdx.x;
    if (i < n) dst[i] = f2bf(src[i]);
}

// W [K x CH] f32 -> Wt [CH x K] bf16 (transposed so B-fragment K is contiguous)
__global__ void cvt_w_kernel(const float* __restrict__ W,
                             unsigned short* __restrict__ Wt, int K) {
    int i = blockIdx.x * blockDim.x + threadIdx.x;
    if (i < K * CH) {
        int k = i / CH, n = i % CH;
        Wt[n * K + k] = f2bf(W[i]);
    }
}

// ---------------- WMMA GEMM: C[N,CH] = A[N,K](bf16) * W[K,CH] (Wt = W^T bf16) ----
// one wave -> one 16x16 tile; blockDim (32,4) = 4 waves; grid (N/16, CH/64)
__global__ void gemm_bf16_wmma_kernel(const unsigned short* __restrict__ A,
                                      const unsigned short* __restrict__ Bt,
                                      float* __restrict__ C, int K) {
    int lane = threadIdx.x;                     // 0..31
    int m0 = blockIdx.x * 16;
    int n0 = (blockIdx.y * 4 + threadIdx.y) * 16;
    int r  = lane & 15;
    int hi = lane >> 4;                         // 0 or 1 (K-half select)
    const unsigned int* ap =
        (const unsigned int*)(A + (size_t)(m0 + r) * K);   // A row m0+r
    const unsigned int* bp =
        (const unsigned int*)(Bt + (size_t)(n0 + r) * K);  // W col n0+r (row of Wt)
    int pbase = hi * 4;                          // dword-pair base within 16-wide K window

    v8f acc = {};
    for (int k0 = 0; k0 < K; k0 += 32) {
        union { unsigned int u[8]; v16bf v; } fa, fb;
        int kp = k0 >> 1;                        // dword index of k0
        #pragma unroll
        for (int i = 0; i < 4; ++i) {
            fa.u[i]     = ap[kp + pbase + i];        // K = kbase + 2i (+1)
            fa.u[i + 4] = ap[kp + 8 + pbase + i];    // K = 16 + kbase + 2i (+1)
            fb.u[i]     = bp[kp + pbase + i];
            fb.u[i + 4] = bp[kp + 8 + pbase + i];
        }
        acc = __builtin_amdgcn_wmma_f32_16x16x32_bf16(
            false, fa.v, false, fb.v, (short)0, acc, false, false);
    }
    // D: VGPR r -> row m0 + r + 8*hi, col n0 + lane%16
    float* out = C + (size_t)(m0 + hi * 8) * CH + n0 + r;
    #pragma unroll
    for (int i = 0; i < 8; ++i) out[(size_t)i * CH] = acc[i];
}

// ---------------- attention coefficients ----------------
__global__ void alpha_kernel(const float* __restrict__ h,
                             const float* __restrict__ as_,
                             const float* __restrict__ ad_,
                             float* __restrict__ asrc,
                             float* __restrict__ adst) {
    int t = blockIdx.x * blockDim.x + threadIdx.x;
    if (t >= N_NODES * HEADS) return;
    int n = t >> 2, hh = t & 3;
    const float* hp  = h + (size_t)n * CH + hh * HID;
    const float* asp = as_ + hh * HID;
    const float* adp = ad_ + hh * HID;
    float s1 = 0.f, s2 = 0.f;
    #pragma unroll 8
    for (int d = 0; d < HID; ++d) { float v = hp[d]; s1 += v * asp[d]; s2 += v * adp[d]; }
    asrc[t] = s1; adst[t] = s2;
}

__device__ __forceinline__ void edge_sd(const int* ei, int E, int e, int& s, int& d) {
    if (e < E) { s = ei[e]; d = ei[E + e]; } else { s = e - E; d = s; }  // self-loops
}

// pass 1: leaky-relu logits + segment max over dst
__global__ void edge_max_kernel(const int* __restrict__ ei, int E, int Etot,
                                const float* __restrict__ asrc,
                                const float* __restrict__ adst,
                                float* __restrict__ ep, float* __restrict__ mbuf) {
    int t = blockIdx.x * blockDim.x + threadIdx.x;
    if (t >= Etot * HEADS) return;
    int e = t >> 2, hh = t & 3, s, d;
    edge_sd(ei, E, e, s, d);
    float v = asrc[s * HEADS + hh] + adst[d * HEADS + hh];
    v = v > 0.f ? v : NEG_SLOPE * v;
    ep[t] = v;
    atomicMax(mbuf + d * HEADS + hh, v);
}

// pass 2: exp(e - max) + segment sum
__global__ void edge_sum_kernel(const int* __restrict__ ei, int E, int Etot,
                                const float* __restrict__ mbuf,
                                float* __restrict__ ep, float* __restrict__ sbuf) {
    int t = blockIdx.x * blockDim.x + threadIdx.x;
    if (t >= Etot * HEADS) return;
    int e = t >> 2, hh = t & 3, s, d;
    edge_sd(ei, E, e, s, d);
    float x = __expf(ep[t] - mbuf[d * HEADS + hh]);
    ep[t] = x;
    atomicAdd(sbuf + d * HEADS + hh, x);
}

// pass 3: weighted message aggregation (one block per edge, 256 channels)
__global__ void edge_agg_kernel(const int* __restrict__ ei, int E,
                                const float* __restrict__ h,
                                const float* __restrict__ ep,
                                const float* __restrict__ sbuf,
                                float* __restrict__ agg) {
    int e = blockIdx.x;
    int c = threadIdx.x;                 // 0..255
    int s, d;
    edge_sd(ei, E, e, s, d);
    int hh = c >> 6;
    float alpha = ep[e * HEADS + hh] / sbuf[d * HEADS + hh];
    atomicAdd(agg + (size_t)d * CH + c, h[(size_t)s * CH + c] * alpha);
}

// bias + ELU, write f32 (in place) and bf16 copy for next layer's GEMM
__global__ void bias_elu_kernel(float* __restrict__ agg, const float* __restrict__ b,
                                unsigned short* __restrict__ xb) {
    int i = blockIdx.x * blockDim.x + threadIdx.x;
    if (i >= N_NODES * CH) return;
    float v = agg[i] + b[i & (CH - 1)];
    v = v > 0.f ? v : __expf(v) - 1.f;
    agg[i] = v;
    xb[i] = f2bf(v);
}

// ---------------- pooling + head ----------------
__global__ void pool_kernel(const float* __restrict__ x, const int* __restrict__ batch,
                            float* __restrict__ pool, float* __restrict__ cnt) {
    int i = blockIdx.x * blockDim.x + threadIdx.x;
    if (i >= N_NODES * CH) return;
    int n = i >> 8, c = i & 255;
    int g = batch[n];
    atomicAdd(pool + g * CH + c, x[i]);
    if (c == 0) atomicAdd(cnt + g, 1.0f);
}

__global__ void final_kernel(const float* __restrict__ pool, const float* __restrict__ cnt,
                             const float* __restrict__ Wf, const float* __restrict__ bfv,
                             float* __restrict__ out) {
    __shared__ float gs[CH];
    __shared__ float red[OUT_DIM];
    int g = blockIdx.x, o = threadIdx.x;
    float c = cnt[g]; c = c > 1.f ? c : 1.f;
    for (int k = o; k < CH; k += OUT_DIM) gs[k] = pool[g * CH + k] / c;
    __syncthreads();
    float y = bfv[o];
    for (int k = 0; k < CH; ++k) y += gs[k] * Wf[k * OUT_DIM + o];
    red[o] = y * y;
    __syncthreads();
    for (int st = OUT_DIM / 2; st > 0; st >>= 1) {
        if (o < st) red[o] += red[o + st];
        __syncthreads();
    }
    float norm = sqrtf(red[0]);
    norm = norm > 1e-12f ? norm : 1e-12f;
    out[g * OUT_DIM + o] = y / norm;
}

// ---------------- driver ----------------
extern "C" void kernel_launch(void* const* d_in, const int* in_sizes, int n_in,
                              void* d_out, int out_size, void* d_ws, size_t ws_size,
                              hipStream_t stream) {
    const float* x0    = (const float*)d_in[0];
    const int*   ei    = (const int*)d_in[1];
    const int*   batch = (const int*)d_in[2];
    const float* W[3]  = {(const float*)d_in[3],  (const float*)d_in[7],  (const float*)d_in[11]};
    const float* AS[3] = {(const float*)d_in[4],  (const float*)d_in[8],  (const float*)d_in[12]};
    const float* AD[3] = {(const float*)d_in[5],  (const float*)d_in[9],  (const float*)d_in[13]};
    const float* B[3]  = {(const float*)d_in[6],  (const float*)d_in[10], (const float*)d_in[14]};
    const float* Wf  = (const float*)d_in[15];
    const float* bfv = (const float*)d_in[16];

    const int E    = in_sizes[1] / 2;      // 800000
    const int Etot = E + N_NODES;          // + self loops

    char* ws = (char*)d_ws;
    size_t off = 0;
    auto alloc = [&](size_t bytes) -> void* {
        void* p = ws + off;
        off = (off + bytes + 255) & ~(size_t)255;
        return p;
    };
    unsigned short* xb  = (unsigned short*)alloc((size_t)N_NODES * CH * 2);
    unsigned short* Wt  = (unsigned short*)alloc((size_t)CH * CH * 2);
    float* h    = (float*)alloc((size_t)N_NODES * CH * 4);
    float* agg  = (float*)alloc((size_t)N_NODES * CH * 4);
    float* asrc = (float*)alloc((size_t)N_NODES * HEADS * 4);
    float* adst = (float*)alloc((size_t)N_NODES * HEADS * 4);
    float* mbuf = (float*)alloc((size_t)N_NODES * HEADS * 4);
    float* sbuf = (float*)alloc((size_t)N_NODES * HEADS * 4);
    float* ep   = (float*)alloc((size_t)Etot * HEADS * 4);
    float* pool = (float*)alloc((size_t)N_GRAPHS * CH * 4);
    float* cnt  = (float*)alloc((size_t)N_GRAPHS * 4);

    auto cdiv = [](int a, int b) { return (a + b - 1) / b; };

    // layer 0 input -> bf16
    cvt_bf16_kernel<<<cdiv(N_NODES * 128, 256), 256, 0, stream>>>(x0, xb, N_NODES * 128);

    int K = 128;
    for (int l = 0; l < 3; ++l) {
        cvt_w_kernel<<<cdiv(K * CH, 256), 256, 0, stream>>>(W[l], Wt, K);
        gemm_bf16_wmma_kernel<<<dim3(N_NODES / 16, CH / 64), dim3(32, 4), 0, stream>>>(
            xb, Wt, h, K);
        alpha_kernel<<<cdiv(N_NODES * HEADS, 256), 256, 0, stream>>>(h, AS[l], AD[l], asrc, adst);

        fill_f32_kernel<<<cdiv(N_NODES * HEADS, 256), 256, 0, stream>>>(mbuf, -1e30f, N_NODES * HEADS);
        fill_f32_kernel<<<cdiv(N_NODES * HEADS, 256), 256, 0, stream>>>(sbuf, 0.f, N_NODES * HEADS);
        fill_f32_kernel<<<cdiv(N_NODES * CH, 256), 256, 0, stream>>>(agg, 0.f, N_NODES * CH);

        edge_max_kernel<<<cdiv(Etot * HEADS, 256), 256, 0, stream>>>(ei, E, Etot, asrc, adst, ep, mbuf);
        edge_sum_kernel<<<cdiv(Etot * HEADS, 256), 256, 0, stream>>>(ei, E, Etot, mbuf, ep, sbuf);
        edge_agg_kernel<<<Etot, CH, 0, stream>>>(ei, E, h, ep, sbuf, agg);

        bias_elu_kernel<<<cdiv(N_NODES * CH, 256), 256, 0, stream>>>(agg, B[l], xb);
        K = CH;
    }

    fill_f32_kernel<<<cdiv(N_GRAPHS * CH, 256), 256, 0, stream>>>(pool, 0.f, N_GRAPHS * CH);
    fill_f32_kernel<<<1, 256, 0, stream>>>(cnt, 0.f, N_GRAPHS);
    pool_kernel<<<cdiv(N_NODES * CH, 256), 256, 0, stream>>>(agg, batch, pool, cnt);
    final_kernel<<<N_GRAPHS, OUT_DIM, 0, stream>>>(pool, cnt, Wf, bfv, (float*)d_out);
}
